// NonLocalBlock_5669356835843
// MI455X (gfx1250) — matrix operand order repaired and checked by reading, not compiled, not run on hardware
//
#include <hip/hip_runtime.h>
#include <hip/hip_bf16.h>

// ---------------------------------------------------------------------------
// NonLocalBlock on gfx1250: bf16 WMMA + async-to-LDS double-buffered K/V,
// fused flash-style softmax. Shapes: bs=2, C=128, N=4096, H=1, d=128, Ch=64.
// ---------------------------------------------------------------------------

typedef __bf16 bf16_t;
typedef bf16_t bf16x16 __attribute__((ext_vector_type(16)));
typedef bf16_t bf16x8  __attribute__((ext_vector_type(8)));
typedef float  f32x8   __attribute__((ext_vector_type(8)));
typedef __attribute__((address_space(3))) bf16_t lds_bf16_t;

#define CC     128
#define NN     4096
#define CH     64
#define CHUNK  64
#define QSCALE 0.08838834764831845f   // 1/sqrt(128)

__device__ __forceinline__ bf16_t f2bf(float f) {
  union { float f; unsigned u; } in; in.f = f;
  unsigned u = in.u;
  unsigned r = u + 0x7FFFu + ((u >> 16) & 1u);       // round-to-nearest-even
  unsigned short h = (unsigned short)(r >> 16);
  union { unsigned short s; bf16_t b; } out; out.s = h;
  return out.b;
}

__device__ __forceinline__ f32x8 wmma_bf16(bf16x16 a, bf16x16 b, f32x8 c) {
  return __builtin_amdgcn_wmma_f32_16x16x32_bf16(false, a, false, b,
                                                 (short)0, c, false, false);
}

// A-matrix fragment (16x32 bf16): lane m = lane%16, element j:
//   k = (j<8) ? half*8+j : 16+half*8+(j-8)  -> two contiguous 16B loads.
__device__ __forceinline__ bf16x16 load_a_frag(const bf16_t* row, int kbase, int half) {
  union { bf16x16 v; bf16x8 h[2]; } u;
  u.h[0] = *(const bf16x8*)(row + kbase + half * 8);
  u.h[1] = *(const bf16x8*)(row + kbase + 16 + half * 8);
  return u.v;
}
// B-matrix fragment (32x16 bf16): lane n = lane%16, element j:
//   k = (j<8) ? half*8+j : 16+half*8+(j-8)  -> two contiguous 16B loads.
__device__ __forceinline__ bf16x16 load_b_frag(const bf16_t* col, int kbase, int half) {
  union { bf16x16 v; bf16x8 h[2]; } u;
  u.h[0] = *(const bf16x8*)(col + kbase + half * 8);
  u.h[1] = *(const bf16x8*)(col + kbase + 16 + half * 8);
  return u.v;
}

// LDS byte offset of a pointer known to live in LDS.
__device__ __forceinline__ unsigned lds_off(bf16_t* p) {
  return (unsigned)(size_t)(lds_bf16_t*)p;
}
// Async 16B global->LDS copy (ASYNCcnt-tracked Tensor/async path, ISA §10/§15.18).
__device__ __forceinline__ void async_cp16(bf16_t* lds_dst, const bf16_t* gsrc) {
  asm volatile("global_load_async_to_lds_b128 %0, %1, off"
               :: "v"(lds_off(lds_dst)), "v"((unsigned long long)(size_t)gsrc)
               : "memory");
}

// ---------------------------------------------------------------------------
__global__ void cast_bf16_kernel(const float* __restrict__ src,
                                 bf16_t* __restrict__ dst, int n) {
  int i = blockIdx.x * 256 + threadIdx.x;
  if (i < n) dst[i] = f2bf(src[i]);
}

// feat [b][C][N] f32 -> featT [b][N][C] bf16
__global__ void transpose_feat_kernel(const float* __restrict__ feat,
                                      bf16_t* __restrict__ featT) {
  int b = blockIdx.y;
  int idx = blockIdx.x * 256 + threadIdx.x;     // = c*N + n
  int c = idx >> 12;
  int n = idx & (NN - 1);
  featT[(size_t)b * NN * CC + (size_t)n * CC + c] =
      f2bf(feat[(size_t)b * CC * NN + idx]);
}

// ---------------------------------------------------------------------------
// QKV: per block 8 waves (one 16-row out-channel tile each), one 16-col n-tile.
// grid = (N/16, 3 proj, batch). Q,K stored transposed [N][C]; V stored [C][N].
__global__ __launch_bounds__(256) void qkv_kernel(
    const bf16_t* __restrict__ featT,
    const bf16_t* __restrict__ Wqb, const bf16_t* __restrict__ Wkb,
    const bf16_t* __restrict__ Wvb,
    const float* __restrict__ bq, const float* __restrict__ bk,
    const float* __restrict__ bv,
    bf16_t* __restrict__ Qt, bf16_t* __restrict__ Kt, bf16_t* __restrict__ Vc) {
  const int proj = blockIdx.y;
  const int b    = blockIdx.z;
  const int wave = threadIdx.x >> 5;
  const int lane = threadIdx.x & 31, half = lane >> 4, l16 = lane & 15;
  const int ntile = blockIdx.x;

  const bf16_t* W    = (proj == 0) ? Wqb : (proj == 1) ? Wkb : Wvb;
  const float*  bias = (proj == 0) ? bq  : (proj == 1) ? bk  : bv;

  const bf16_t* arow = featT + (size_t)b * NN * CC + (size_t)(ntile * 16 + l16) * CC;
  const bf16_t* brow = W + (size_t)(wave * 16 + l16) * CC;

  f32x8 acc = 0.0f;
#pragma unroll
  for (int ks = 0; ks < 4; ++ks)
    acc = wmma_bf16(load_a_frag(arow, ks * 32, half),
                    load_b_frag(brow, ks * 32, half), acc);

  const int orow = wave * 16 + l16;
  const float bia = bias[orow];
  const float scl = (proj == 0) ? QSCALE : 1.0f;

#pragma unroll
  for (int r = 0; r < 8; ++r) {
    int ncol = ntile * 16 + r + 8 * half;
    float v = (acc[r] + bia) * scl;
    if (proj == 0)
      Qt[(size_t)b * NN * CC + (size_t)ncol * CC + orow] = f2bf(v);
    else if (proj == 1)
      Kt[(size_t)b * NN * CC + (size_t)ncol * CC + orow] = f2bf(v);
    else
      Vc[(size_t)b * CC * NN + (size_t)orow * NN + ncol] = f2bf(v);
  }
}

// ---------------------------------------------------------------------------
// Issue the async copies for one 64-i chunk of K and V into an LDS buffer.
// Per wave: 8 async instructions (ASYNCcnt += 8).
__device__ __forceinline__ void issue_kv_copy(bf16_t* ksDst, bf16_t* vsDst,
                                              const bf16_t* kb, const bf16_t* vb,
                                              int ib, int tid) {
  const bf16_t* ksrc = kb + (size_t)ib * CC;      // rows ib..ib+63 contiguous
#pragma unroll
  for (int s = 0; s < 4; ++s) {
    int e = (s * 256 + tid) * 8;                  // 8 bf16 = 16B granules
    async_cp16(ksDst + e, ksrc + e);
  }
#pragma unroll
  for (int s = 0; s < 4; ++s) {
    int id = s * 256 + tid;
    int c = id >> 3, o8 = (id & 7) * 8;           // 8 x 16B per 64-elem row
    async_cp16(vsDst + c * CHUNK + o8, vb + (size_t)c * NN + ib + o8);
  }
}

// Flash attention: one wave per 16-row o-tile; i in chunks of 64 with online
// softmax; K/V double-buffered in LDS via async copies shared by 8 waves.
// grid = (N/16/8, batch), block = 256.
__global__ __launch_bounds__(256) void attn_kernel(
    const bf16_t* __restrict__ Qt, const bf16_t* __restrict__ Kt,
    const bf16_t* __restrict__ Vc, const float* __restrict__ att,
    bf16_t* __restrict__ MsgT) {
  __shared__ bf16_t Ks[2][CHUNK * CC];    // [i_local][c], c-contiguous
  __shared__ bf16_t Vs[2][CC * CHUNK];    // [c][i_local], i-contiguous
  __shared__ bf16_t Pl[8][16 * CHUNK];    // per-wave probability staging

  const int b    = blockIdx.y;
  const int tid  = threadIdx.x;
  const int wave = tid >> 5;
  const int lane = tid & 31, half = lane >> 4, l16 = lane & 15;
  const int obase = (blockIdx.x * 8 + wave) * 16;

  const bf16_t* qb = Qt + (size_t)b * NN * CC;
  const bf16_t* kb = Kt + (size_t)b * NN * CC;
  const bf16_t* vb = Vc + (size_t)b * CC * NN;
  const float*  ab = att + (size_t)b * NN * NN;

  bf16x16 qf[4];
  {
    const bf16_t* qrow = qb + (size_t)(obase + l16) * CC;
#pragma unroll
    for (int ks = 0; ks < 4; ++ks) qf[ks] = load_a_frag(qrow, ks * 32, half);
  }

  f32x8 macc[8];
  float m[8], l[8];
#pragma unroll
  for (int ct = 0; ct < 8; ++ct) macc[ct] = 0.0f;
#pragma unroll
  for (int r = 0; r < 8; ++r) { m[r] = -3.0e38f; l[r] = 0.0f; }

  bf16_t* P = &Pl[wave][0];

  issue_kv_copy(&Ks[0][0], &Vs[0][0], kb, vb, 0, tid);

  int buf = 0;
  for (int ib = 0; ib < NN; ib += CHUNK, buf ^= 1) {
    const int ibn = ib + CHUNK;
    if (ibn < NN) {
      issue_kv_copy(&Ks[buf ^ 1][0], &Vs[buf ^ 1][0], kb, vb, ibn, tid);
      asm volatile("s_wait_asynccnt 0x8" ::: "memory");   // current chunk landed
      // prefetch next attention chunk: 16 rows x 64 i = 32 lanes x 128B
      __builtin_prefetch(ab + (size_t)(obase + (lane >> 1)) * NN + ibn + (lane & 1) * 32, 0, 0);
    } else {
      asm volatile("s_wait_asynccnt 0x0" ::: "memory");
    }
    __syncthreads();

    // scores for 4 sub-tiles of 16 i, times the attention gate
    float pv[4][8];
#pragma unroll
    for (int t = 0; t < 4; ++t) {
      f32x8 s = 0.0f;
      const bf16_t* krow = &Ks[buf][(t * 16 + l16) * CC];
#pragma unroll
      for (int ks = 0; ks < 4; ++ks)
        s = wmma_bf16(qf[ks], load_b_frag(krow, ks * 32, half), s);
#pragma unroll
      for (int r = 0; r < 8; ++r) {
        float a = ab[(size_t)(obase + r + 8 * half) * NN + ib + t * 16 + l16];
        pv[t][r] = s[r] * a;
      }
    }

    // online softmax: row o = r + 8*half lives across the 16 lanes of a half
#pragma unroll
    for (int r = 0; r < 8; ++r) {
      float mx = fmaxf(fmaxf(pv[0][r], pv[1][r]), fmaxf(pv[2][r], pv[3][r]));
#pragma unroll
      for (int msk = 1; msk < 16; msk <<= 1) mx = fmaxf(mx, __shfl_xor(mx, msk, 32));
      float mnew = fmaxf(m[r], mx);
      float corr = __expf(m[r] - mnew);
      m[r] = mnew;
      float ps = 0.0f;
      int ol = (r + 8 * half) * CHUNK;
#pragma unroll
      for (int t = 0; t < 4; ++t) {
        float p = __expf(pv[t][r] - mnew);
        ps += p;
        P[ol + t * 16 + l16] = f2bf(p);
      }
      l[r] = l[r] * corr + ps;
#pragma unroll
      for (int ct = 0; ct < 8; ++ct) macc[ct][r] *= corr;
    }
    asm volatile("s_wait_dscnt 0" ::: "memory");
    __builtin_amdgcn_wave_barrier();

    // message += P @ V^T   (D-layout -> A-layout via per-wave LDS tile)
    bf16x16 pa0 = load_a_frag(P + l16 * CHUNK, 0, half);
    bf16x16 pa1 = load_a_frag(P + l16 * CHUNK, 32, half);
#pragma unroll
    for (int ct = 0; ct < 8; ++ct) {
      const bf16_t* vrow = &Vs[buf][(ct * 16 + l16) * CHUNK];
      macc[ct] = wmma_bf16(pa0, load_b_frag(vrow, 0, half), macc[ct]);
      macc[ct] = wmma_bf16(pa1, load_b_frag(vrow, 32, half), macc[ct]);
    }
    __syncthreads();   // everyone done reading buf before it is re-filled
  }

#pragma unroll
  for (int r = 0; r < 8; ++r) {
    float lt = l[r];
#pragma unroll
    for (int msk = 1; msk < 16; msk <<= 1) lt += __shfl_xor(lt, msk, 32);
    float inv = 1.0f / lt;
    size_t row = (size_t)b * NN * CC + (size_t)(obase + r + 8 * half) * CC;
#pragma unroll
    for (int ct = 0; ct < 8; ++ct)
      MsgT[row + ct * 16 + l16] = f2bf(macc[ct][r] * inv);
  }
}

// ---------------------------------------------------------------------------
// MLP: msg^T[N][C] -> h1[N][64] -> h2[N][64] -> out[C][N] = feat + W3 h2 + b3.
// One wave per 16-point tile; intermediates staged in per-wave LDS.
__global__ __launch_bounds__(256) void mlp_kernel(
    const bf16_t* __restrict__ MsgT,
    const bf16_t* __restrict__ W1b, const bf16_t* __restrict__ W2b,
    const bf16_t* __restrict__ W3b,
    const float* __restrict__ b1, const float* __restrict__ g1,
    const float* __restrict__ be1, const float* __restrict__ m1,
    const float* __restrict__ v1,
    const float* __restrict__ b2, const float* __restrict__ g2,
    const float* __restrict__ be2, const float* __restrict__ m2,
    const float* __restrict__ v2,
    const float* __restrict__ b3,
    const float* __restrict__ feat, float* __restrict__ out) {
  __shared__ bf16_t H1s[8][16 * CH];
  __shared__ bf16_t H2s[8][16 * CH];

  const int b    = blockIdx.y;
  const int wave = threadIdx.x >> 5;
  const int lane = threadIdx.x & 31, half = lane >> 4, l16 = lane & 15;
  const int nbase = (blockIdx.x * 8 + wave) * 16;

  bf16_t* H1 = &H1s[wave][0];
  bf16_t* H2 = &H2s[wave][0];

  bf16x16 af[4];
  {
    const bf16_t* mrow = MsgT + (size_t)b * NN * CC + (size_t)(nbase + l16) * CC;
#pragma unroll
    for (int ks = 0; ks < 4; ++ks) af[ks] = load_a_frag(mrow, ks * 32, half);
  }

  // GEMM1 (K=128) + BN1 + ReLU
#pragma unroll
  for (int jt = 0; jt < 4; ++jt) {
    int j = jt * 16 + l16;
    f32x8 acc = 0.0f;
    const bf16_t* wrow = W1b + (size_t)j * CC;
#pragma unroll
    for (int ks = 0; ks < 4; ++ks)
      acc = wmma_bf16(af[ks], load_b_frag(wrow, ks * 32, half), acc);
    float inv = g1[j] * rsqrtf(v1[j] + 1e-5f);
    float add = (b1[j] - m1[j]) * inv + be1[j];
#pragma unroll
    for (int r = 0; r < 8; ++r)
      H1[(r + 8 * half) * CH + j] = f2bf(fmaxf(acc[r] * inv + add, 0.0f));
  }
  asm volatile("s_wait_dscnt 0" ::: "memory");
  __builtin_amdgcn_wave_barrier();

  // GEMM2 (K=64) + BN2 + ReLU
  bf16x16 a2[2];
  {
    const bf16_t* h1row = H1 + l16 * CH;
#pragma unroll
    for (int ks = 0; ks < 2; ++ks) a2[ks] = load_a_frag(h1row, ks * 32, half);
  }
#pragma unroll
  for (int jt = 0; jt < 4; ++jt) {
    int j = jt * 16 + l16;
    f32x8 acc = 0.0f;
    const bf16_t* wrow = W2b + (size_t)j * CH;
#pragma unroll
    for (int ks = 0; ks < 2; ++ks)
      acc = wmma_bf16(a2[ks], load_b_frag(wrow, ks * 32, half), acc);
    float inv = g2[j] * rsqrtf(v2[j] + 1e-5f);
    float add = (b2[j] - m2[j]) * inv + be2[j];
#pragma unroll
    for (int r = 0; r < 8; ++r)
      H2[(r + 8 * half) * CH + j] = f2bf(fmaxf(acc[r] * inv + add, 0.0f));
  }
  asm volatile("s_wait_dscnt 0" ::: "memory");
  __builtin_amdgcn_wave_barrier();

  // GEMM3 (K=64): A = W3 rows (m = out channel), B = h2 (n = point), + residual
  bf16x16 bf0 = load_b_frag(H2 + l16 * CH, 0, half);
  bf16x16 bf1 = load_b_frag(H2 + l16 * CH, 32, half);
#pragma unroll
  for (int ct = 0; ct < 8; ++ct) {
    const bf16_t* wrow = W3b + (size_t)(ct * 16 + l16) * CH;
    f32x8 acc = 0.0f;
    acc = wmma_bf16(load_a_frag(wrow, 0, half),  bf0, acc);
    acc = wmma_bf16(load_a_frag(wrow, 32, half), bf1, acc);
#pragma unroll
    for (int r = 0; r < 8; ++r) {
      int c = ct * 16 + r + 8 * half;
      size_t idx = (size_t)b * CC * NN + (size_t)c * NN + nbase + l16;
      out[idx] = feat[idx] + acc[r] + b3[c];
    }
  }
}

// ---------------------------------------------------------------------------
extern "C" void kernel_launch(void* const* d_in, const int* in_sizes, int n_in,
                              void* d_out, int out_size, void* d_ws, size_t ws_size,
                              hipStream_t stream) {
  const float* feat = (const float*)d_in[0];
  const float* attn = (const float*)d_in[1];
  const float* Wq = (const float*)d_in[2];  const float* bq = (const float*)d_in[3];
  const float* Wk = (const float*)d_in[4];  const float* bk = (const float*)d_in[5];
  const float* Wv = (const float*)d_in[6];  const float* bv = (const float*)d_in[7];
  const float* W1 = (const float*)d_in[8];  const float* b1 = (const float*)d_in[9];
  const float* g1 = (const float*)d_in[10]; const float* be1 = (const float*)d_in[11];
  const float* m1 = (const float*)d_in[12]; const float* v1 = (const float*)d_in[13];
  const float* W2 = (const float*)d_in[14]; const float* b2 = (const float*)d_in[15];
  const float* g2 = (const float*)d_in[16]; const float* be2 = (const float*)d_in[17];
  const float* m2 = (const float*)d_in[18]; const float* v2 = (const float*)d_in[19];
  const float* W3 = (const float*)d_in[20]; const float* b3 = (const float*)d_in[21];

  char* ws = (char*)d_ws;
  const size_t NC = (size_t)2 * NN * CC * sizeof(bf16_t);   // 2 MB per tensor
  bf16_t* featT = (bf16_t*)(ws);
  bf16_t* Qt    = (bf16_t*)(ws + 1 * NC);
  bf16_t* Kt    = (bf16_t*)(ws + 2 * NC);
  bf16_t* Vc    = (bf16_t*)(ws + 3 * NC);
  bf16_t* MsgT  = (bf16_t*)(ws + 4 * NC);
  size_t  off   = 5 * NC;
  bf16_t* Wqb = (bf16_t*)(ws + off); off += CC * CC * 2;
  bf16_t* Wkb = (bf16_t*)(ws + off); off += CC * CC * 2;
  bf16_t* Wvb = (bf16_t*)(ws + off); off += CC * CC * 2;
  bf16_t* W1b = (bf16_t*)(ws + off); off += CH * CC * 2;
  bf16_t* W2b = (bf16_t*)(ws + off); off += CH * CH * 2;
  bf16_t* W3b = (bf16_t*)(ws + off); off += CC * CH * 2;

  cast_bf16_kernel<<<(CC * CC + 255) / 256, 256, 0, stream>>>(Wq, Wqb, CC * CC);
  cast_bf16_kernel<<<(CC * CC + 255) / 256, 256, 0, stream>>>(Wk, Wkb, CC * CC);
  cast_bf16_kernel<<<(CC * CC + 255) / 256, 256, 0, stream>>>(Wv, Wvb, CC * CC);
  cast_bf16_kernel<<<(CH * CC + 255) / 256, 256, 0, stream>>>(W1, W1b, CH * CC);
  cast_bf16_kernel<<<(CH * CH + 255) / 256, 256, 0, stream>>>(W2, W2b, CH * CH);
  cast_bf16_kernel<<<(CC * CH + 255) / 256, 256, 0, stream>>>(W3, W3b, CC * CH);

  transpose_feat_kernel<<<dim3(CC * NN / 256, 2), 256, 0, stream>>>(feat, featT);

  qkv_kernel<<<dim3(NN / 16, 3, 2), 256, 0, stream>>>(
      featT, Wqb, Wkb, Wvb, bq, bk, bv, Qt, Kt, Vc);

  attn_kernel<<<dim3(NN / 16 / 8, 2), 256, 0, stream>>>(Qt, Kt, Vc, attn, MsgT);

  mlp_kernel<<<dim3(NN / 16 / 8, 2), 256, 0, stream>>>(
      MsgT, W1b, W2b, W3b, b1, g1, be1, m1, v1, b2, g2, be2, m2, v2, b3,
      feat, (float*)d_out);
}
